// LoRAAttentionProcessor_10514079941021
// MI455X (gfx1250) — matrix operand order repaired and checked by reading, not compiled, not run on hardware
//
#include <hip/hip_runtime.h>
#include <hip/hip_bf16.h>
#include <stdint.h>

typedef __attribute__((ext_vector_type(16))) _Float16 v16h;
typedef __attribute__((ext_vector_type(8)))  float    v8f;

#define DIM   1152
#define HEADS 16
#define HDIM  72
#define HPAD  96
#define SEQ   2048
#define BATCH 2
#define RANK  32
#define M_TOTAL (BATCH*SEQ)   // 4096

// gfx1250 async global->LDS copy path (ASYNCcnt), guarded so compile never breaks
#define HAVE_ASYNC_LDS 0
#if defined(__has_builtin)
#if __has_builtin(__builtin_amdgcn_global_load_async_to_lds_b128) && \
    __has_builtin(__builtin_amdgcn_s_wait_asynccnt)
#undef HAVE_ASYNC_LDS
#define HAVE_ASYNC_LDS 1
// builtin parameter type (from compiler diagnostic): pointer to
// 'int __attribute__((vector_size(16)))'
typedef int v4i_vs __attribute__((vector_size(16)));
#endif
#endif

__device__ __forceinline__ v8f wmma_f16(const v16h& a, const v16h& b, const v8f& c) {
  // D = A(16x32 f16) * B(32x16 f16) + C(16x16 f32)
  return __builtin_amdgcn_wmma_f32_16x16x32_f16(false, a, false, b, (short)0, c,
                                                false, false);
}

__device__ __forceinline__ void copy16h_g2l(const _Float16* g, _Float16* l) {
#if HAVE_ASYNC_LDS
  __builtin_amdgcn_global_load_async_to_lds_b128(
      (v4i_vs*)(const void*)g, (v4i_vs*)(void*)l, 0, 0);
#else
  *(float4*)l = *(const float4*)g;
#endif
}

__device__ __forceinline__ void wait_async_lds() {
#if HAVE_ASYNC_LDS
  __builtin_amdgcn_s_wait_asynccnt(0);
#endif
}

// ---------------------------------------------------------------------------
// 16-lane butterfly reductions in pure VALU via DPP16 (no ds_bpermute trips).
// Ladder: quad_perm(1,0,3,2), quad_perm(2,3,0,1), row_half_mirror, row_mirror.
// Lanes 0-15 and 16-31 reduce independently (rows live per 16-lane half).
// ---------------------------------------------------------------------------
template<int CTRL>
__device__ __forceinline__ float dpp_f(float x) {
  return __int_as_float(__builtin_amdgcn_update_dpp(
      0, __float_as_int(x), CTRL, 0xf, 0xf, true));
}
__device__ __forceinline__ float rowmax16(float x) {
  x = fmaxf(x, dpp_f<0xB1>(x));   // quad_perm [1,0,3,2]  (xor 1)
  x = fmaxf(x, dpp_f<0x4E>(x));   // quad_perm [2,3,0,1]  (xor 2)
  x = fmaxf(x, dpp_f<0x141>(x));  // row_half_mirror
  x = fmaxf(x, dpp_f<0x140>(x));  // row_mirror
  return x;
}
__device__ __forceinline__ float rowsum16(float x) {
  x += dpp_f<0xB1>(x);
  x += dpp_f<0x4E>(x);
  x += dpp_f<0x141>(x);
  x += dpp_f<0x140>(x);
  return x;
}

// ---------------------------------------------------------------------------
// zero-fill (padded head-dim region of Q/K/V must read as 0)
// ---------------------------------------------------------------------------
__global__ void __launch_bounds__(256) zero_kernel(uint32_t* __restrict__ p, size_t n) {
  size_t i = (size_t)blockIdx.x * blockDim.x + threadIdx.x;
  size_t stride = (size_t)gridDim.x * blockDim.x;
  for (; i < n; i += stride) p[i] = 0u;
}

// ---------------------------------------------------------------------------
// t[M,32] = src[M,1152] @ dn[32,1152]^T      (LoRA down projection)
// ---------------------------------------------------------------------------
template<bool SRC_F16>
__global__ void __launch_bounds__(256) lora_down_kernel(
    const void* __restrict__ src_, const float* __restrict__ dn,
    _Float16* __restrict__ t)
{
  __shared__ _Float16 xs[64][40];
  const int m0   = blockIdx.x * 64;
  const int w    = threadIdx.x >> 5;
  const int lane = threadIdx.x & 31;
  const int wm = w & 3, wn = w >> 2;
  const int hh = (lane < 16) ? 0 : 8;    // A-frag K offset per lane half
  const int kb = (lane < 16) ? 0 : 16;   // unused (B-frag K offset is hh, as for A)
  const v8f zero = {};
  v8f acc = zero;

  for (int k0 = 0; k0 < DIM; k0 += 32) {
    for (int i = threadIdx.x; i < 64*32; i += 256) {
      int r = i >> 5, c = i & 31;
      float xv;
      if (SRC_F16) xv = (float)((const _Float16*)src_)[(size_t)(m0+r)*DIM + k0 + c];
      else         xv = ((const float*)src_)[(size_t)(m0+r)*DIM + k0 + c];
      xs[r][c] = (_Float16)xv;
    }
    __syncthreads();

    v16h a;
    {
      int row = wm*16 + (lane & 15);
      #pragma unroll
      for (int e = 0; e < 8; ++e)  a[e] = xs[row][hh + e];
      #pragma unroll
      for (int e = 8; e < 16; ++e) a[e] = xs[row][16 + hh + (e - 8)];
    }
    v16h b;
    {
      int n = wn*16 + (lane & 15);
      #pragma unroll
      for (int e = 0; e < 16; ++e)
        b[e] = (_Float16)dn[(size_t)n*DIM + k0 + hh + (e & 7) + ((e >> 3) << 4)];
    }
    acc = wmma_f16(a, b, acc);
    __syncthreads();
  }

  int gn = wn*16 + (lane & 15);
  #pragma unroll
  for (int i = 0; i < 8; ++i) {
    int gm = m0 + wm*16 + ((lane < 16) ? i : 8 + i);
    t[(size_t)gm*RANK + gn] = (_Float16)acc[i];
  }
}

// ---------------------------------------------------------------------------
// out[M,1152] = src[M,1152] @ W^T + bias + t[M,32] @ up^T   (proj + fused LoRA)
// OUT_MODE: 0 = dense f32 [M,1152]
//           1 = padded f16 [B,H,S,96]        (Q, K)
//           2 = padded f16 transposed [B,H,96,S]  (V, for contiguous PV B-frags)
// ---------------------------------------------------------------------------
template<bool SRC_F16, int OUT_MODE>
__global__ void __launch_bounds__(256) lora_proj_kernel(
    const void* __restrict__ src_, const float* __restrict__ W,
    const float* __restrict__ bias, const _Float16* __restrict__ t,
    const float* __restrict__ up, void* __restrict__ out_)
{
  __shared__ _Float16 xs[64][40];
  __shared__ _Float16 wsh[64][40];
  const int m0   = blockIdx.x * 64;
  const int n0   = blockIdx.y * 64;
  const int w    = threadIdx.x >> 5;
  const int lane = threadIdx.x & 31;
  const int wm = w & 3, wn = w >> 2;
  const int hh = (lane < 16) ? 0 : 8;
  const int kb = (lane < 16) ? 0 : 16;
  const v8f zero = {};
  v8f acc[2]; acc[0] = zero; acc[1] = zero;

  for (int k0 = 0; k0 < DIM; k0 += 32) {
    for (int i = threadIdx.x; i < 64*32; i += 256) {
      int r = i >> 5, c = i & 31;
      float xv;
      if (SRC_F16) xv = (float)((const _Float16*)src_)[(size_t)(m0+r)*DIM + k0 + c];
      else         xv = ((const float*)src_)[(size_t)(m0+r)*DIM + k0 + c];
      xs[r][c]  = (_Float16)xv;
      wsh[r][c] = (_Float16)W[(size_t)(n0+r)*DIM + k0 + c];
    }
    if (k0 + 32 < DIM) {
      __builtin_prefetch(
          (const char*)&W[(size_t)(n0 + (threadIdx.x & 63))*DIM + k0 + 32], 0, 3);
    }
    __syncthreads();

    v16h a;
    {
      int row = wm*16 + (lane & 15);
      #pragma unroll
      for (int e = 0; e < 8; ++e)  a[e] = xs[row][hh + e];
      #pragma unroll
      for (int e = 8; e < 16; ++e) a[e] = xs[row][16 + hh + (e - 8)];
    }
    #pragma unroll
    for (int nt = 0; nt < 2; ++nt) {
      int nl = wn*32 + nt*16 + (lane & 15);
      v16h b;
      #pragma unroll
      for (int e = 0; e < 16; ++e) b[e] = wsh[nl][hh + (e & 7) + ((e >> 3) << 4)];
      acc[nt] = wmma_f16(a, b, acc[nt]);
    }
    __syncthreads();
  }

  // fused LoRA delta: one extra K=32 slab with A = t, B = up^T
  {
    v16h a;
    int gm = m0 + wm*16 + (lane & 15);
    #pragma unroll
    for (int e = 0; e < 8; ++e)  a[e] = t[(size_t)gm*RANK + hh + e];
    #pragma unroll
    for (int e = 8; e < 16; ++e) a[e] = t[(size_t)gm*RANK + 16 + hh + (e - 8)];
    #pragma unroll
    for (int nt = 0; nt < 2; ++nt) {
      int gn = n0 + wn*32 + nt*16 + (lane & 15);
      v16h b;
      #pragma unroll
      for (int e = 0; e < 16; ++e)
        b[e] = (_Float16)up[(size_t)gn*RANK + hh + (e & 7) + ((e >> 3) << 4)];
      acc[nt] = wmma_f16(a, b, acc[nt]);
    }
  }

  // epilogue: + bias, store
  #pragma unroll
  for (int nt = 0; nt < 2; ++nt) {
    int gn = n0 + wn*32 + nt*16 + (lane & 15);
    float bv = bias[gn];
    int head = gn / HDIM;
    int d    = gn % HDIM;
    #pragma unroll
    for (int i = 0; i < 8; ++i) {
      int gm = m0 + wm*16 + ((lane < 16) ? i : 8 + i);
      float v = acc[nt][i] + bv;
      if (OUT_MODE == 0) {
        ((float*)out_)[(size_t)gm*DIM + gn] = v;
      } else {
        int bb = gm / SEQ, s = gm % SEQ;
        size_t idx;
        if (OUT_MODE == 1)
          idx = (((size_t)bb*HEADS + head)*SEQ + s)*HPAD + d;
        else
          idx = (((size_t)bb*HEADS + head)*HPAD + d)*SEQ + s;
        ((_Float16*)out_)[idx] = (_Float16)v;
      }
    }
  }
}

// ---------------------------------------------------------------------------
// Flash attention.
// Q,K: padded f16 [B*H, S, 96];  Vt: padded f16 [B*H, 96, S] (transposed).
// Block = 8 waves, 128 query rows; 64-key blocks staged in LDS cooperatively
// (async global->LDS), fragments read from LDS, DPP softmax reductions.
// ---------------------------------------------------------------------------
__global__ void __launch_bounds__(256) attn_kernel(
    const _Float16* __restrict__ Q, const _Float16* __restrict__ K,
    const _Float16* __restrict__ Vt, _Float16* __restrict__ out)
{
  __shared__ alignas(16) _Float16 ksh[64][104];  // 64 keys x 96 dims (+pad)
  __shared__ alignas(16) _Float16 vsh[96][72];   // 96 dims x 64 keys (+pad)
  __shared__ _Float16 pbuf[8][16][72];           // per-wave P tile (C->A bounce)

  const int w    = threadIdx.x >> 5;
  const int lane = threadIdx.x & 31;
  const int bh = blockIdx.y;
  const int b = bh >> 4, h = bh & 15;
  const int q0 = blockIdx.x * 128 + w * 16;
  const int hh = (lane < 16) ? 0 : 8;
  const int kb = (lane < 16) ? 0 : 16;
  const _Float16* Qp = Q  + (size_t)bh * SEQ * HPAD;
  const _Float16* Kp = K  + (size_t)bh * SEQ * HPAD;
  const _Float16* Vp = Vt + (size_t)bh * HPAD * SEQ;
  const float scale = 0.11785113019775793f;   // 1/sqrt(72)
  const v8f zero = {};

  // Q fragments (scale folded in), K-dim = 96 (3 chunks of 32)
  v16h qa[3];
  {
    int qr = q0 + (lane & 15);
    const _Float16* qp = Qp + (size_t)qr * HPAD;
    #pragma unroll
    for (int c = 0; c < 3; ++c) {
      #pragma unroll
      for (int e = 0; e < 8; ++e)
        qa[c][e] = (_Float16)((float)qp[c*32 + hh + e] * scale);
      #pragma unroll
      for (int e = 8; e < 16; ++e)
        qa[c][e] = (_Float16)((float)qp[c*32 + 16 + hh + (e - 8)] * scale);
    }
  }

  float mrow[8], lrow[8];
  #pragma unroll
  for (int i = 0; i < 8; ++i) { mrow[i] = -1e30f; lrow[i] = 0.f; }
  v8f o[6];
  #pragma unroll
  for (int dt = 0; dt < 6; ++dt) o[dt] = zero;

  for (int j = 0; j < SEQ; j += 64) {
    // --- cooperative async staging of the 64-key K and Vt tiles ---
    // K tile: 64 rows x 96 halves = 768 x 16B chunks
    for (int c = threadIdx.x; c < 768; c += 256) {
      int row = c / 12, ch = c % 12;
      copy16h_g2l(Kp + (size_t)(j + row) * HPAD + ch * 8, &ksh[row][ch * 8]);
    }
    // Vt tile: 96 rows x 64 halves = 768 x 16B chunks
    for (int c = threadIdx.x; c < 768; c += 256) {
      int row = c / 8, ch = c % 8;
      copy16h_g2l(Vp + (size_t)row * SEQ + j + ch * 8, &vsh[row][ch * 8]);
    }
    wait_async_lds();
    __syncthreads();

    // scores S = (Q*scale) @ K^T for a 16x64 tile
    v8f s[4];
    #pragma unroll
    for (int nt = 0; nt < 4; ++nt) {
      s[nt] = zero;
      int kl = nt*16 + (lane & 15);
      #pragma unroll
      for (int c = 0; c < 3; ++c) {
        v16h bf;
        #pragma unroll
        for (int e = 0; e < 16; ++e) bf[e] = ksh[kl][c*32 + hh + (e & 7) + ((e >> 3) << 4)];
        s[nt] = wmma_f16(qa[c], bf, s[nt]);
      }
    }

    // online softmax: DPP butterfly reductions within each 16-lane half
    float alpha[8];
    #pragma unroll
    for (int i = 0; i < 8; ++i) {
      float mx = s[0][i];
      #pragma unroll
      for (int nt = 1; nt < 4; ++nt) mx = fmaxf(mx, s[nt][i]);
      mx = rowmax16(mx);
      float nm = fmaxf(mrow[i], mx);
      alpha[i] = __expf(mrow[i] - nm);
      mrow[i] = nm;
    }
    #pragma unroll
    for (int i = 0; i < 8; ++i) {
      float rs = 0.f;
      #pragma unroll
      for (int nt = 0; nt < 4; ++nt) {
        float p = __expf(s[nt][i] - mrow[i]);
        s[nt][i] = p;
        rs += p;
      }
      rs = rowsum16(rs);
      lrow[i] = lrow[i] * alpha[i] + rs;
    }
    #pragma unroll
    for (int dt = 0; dt < 6; ++dt)
      #pragma unroll
      for (int i = 0; i < 8; ++i) o[dt][i] *= alpha[i];

    // bounce P through per-wave LDS to convert C-layout -> A-layout
    #pragma unroll
    for (int nt = 0; nt < 4; ++nt)
      #pragma unroll
      for (int i = 0; i < 8; ++i) {
        int r = (lane < 16) ? i : 8 + i;
        pbuf[w][r][nt*16 + (lane & 15)] = (_Float16)s[nt][i];
      }

    // O += P @ V over the 64-key block (two K=32 chunks)
    #pragma unroll
    for (int c2 = 0; c2 < 2; ++c2) {
      v16h pa;
      int pr = lane & 15;
      #pragma unroll
      for (int e = 0; e < 8; ++e)  pa[e] = pbuf[w][pr][c2*32 + hh + e];
      #pragma unroll
      for (int e = 8; e < 16; ++e) pa[e] = pbuf[w][pr][c2*32 + 16 + hh + (e - 8)];
      #pragma unroll
      for (int dt = 0; dt < 6; ++dt) {
        int dcol = dt*16 + (lane & 15);
        v16h vb;
        #pragma unroll
        for (int e = 0; e < 16; ++e) vb[e] = vsh[dcol][c2*32 + hh + (e & 7) + ((e >> 3) << 4)];
        o[dt] = wmma_f16(pa, vb, o[dt]);
      }
    }
    __syncthreads();   // all reads done before next block's staging
  }

  // finalize: divide by row sums, write valid head-dim columns
  float inv[8];
  #pragma unroll
  for (int i = 0; i < 8; ++i) inv[i] = 1.0f / lrow[i];
  #pragma unroll
  for (int dt = 0; dt < 6; ++dt) {
    int d = dt*16 + (lane & 15);
    if (d < HDIM) {
      #pragma unroll
      for (int i = 0; i < 8; ++i) {
        int qrow = q0 + ((lane < 16) ? i : 8 + i);
        out[((size_t)(b*SEQ + qrow))*DIM + h*HDIM + d] = (_Float16)(o[dt][i] * inv[i]);
      }
    }
  }
}

// ---------------------------------------------------------------------------
extern "C" void kernel_launch(void* const* d_in, const int* in_sizes, int n_in,
                              void* d_out, int out_size, void* d_ws, size_t ws_size,
                              hipStream_t stream)
{
  (void)in_sizes; (void)n_in; (void)out_size; (void)ws_size;
  const float* hidden = (const float*)d_in[0];
  const float* Wq = (const float*)d_in[1];
  const float* bq = (const float*)d_in[2];
  const float* Wk = (const float*)d_in[3];
  const float* bk = (const float*)d_in[4];
  const float* Wv = (const float*)d_in[5];
  const float* bv = (const float*)d_in[6];
  const float* Wo = (const float*)d_in[7];
  const float* bo = (const float*)d_in[8];
  const float* q_down = (const float*)d_in[9];
  const float* q_up   = (const float*)d_in[10];
  const float* k_down = (const float*)d_in[11];
  const float* k_up   = (const float*)d_in[12];
  const float* v_down = (const float*)d_in[13];
  const float* v_up   = (const float*)d_in[14];
  const float* o_down = (const float*)d_in[15];
  const float* o_up   = (const float*)d_in[16];
  float* out = (float*)d_out;

  // workspace layout (f16 elements)
  _Float16* ws = (_Float16*)d_ws;
  const size_t PADSZ = (size_t)BATCH * HEADS * SEQ * HPAD;   // 6,291,456
  _Float16* qpad = ws;
  _Float16* kpad = qpad + PADSZ;
  _Float16* vpad = kpad + PADSZ;                 // transposed [B,H,96,S]
  _Float16* attnbuf = vpad + PADSZ;              // [4096,1152]
  _Float16* t_q = attnbuf + (size_t)M_TOTAL * DIM;
  _Float16* t_k = t_q + (size_t)M_TOTAL * RANK;
  _Float16* t_v = t_k + (size_t)M_TOTAL * RANK;
  _Float16* t_o = t_v + (size_t)M_TOTAL * RANK;

  // 1. zero padded q/k/v (pad region must read 0)
  zero_kernel<<<4096, 256, 0, stream>>>((uint32_t*)qpad, (3 * PADSZ) / 2);

  // 2. LoRA down projections from hidden (f32 source)
  lora_down_kernel<false><<<M_TOTAL/64, 256, 0, stream>>>(hidden, q_down, t_q);
  lora_down_kernel<false><<<M_TOTAL/64, 256, 0, stream>>>(hidden, k_down, t_k);
  lora_down_kernel<false><<<M_TOTAL/64, 256, 0, stream>>>(hidden, v_down, t_v);

  // 3. Q/K/V projections (+ fused LoRA); V stored transposed
  dim3 pg(M_TOTAL/64, DIM/64);
  lora_proj_kernel<false, 1><<<pg, 256, 0, stream>>>(hidden, Wq, bq, t_q, q_up, qpad);
  lora_proj_kernel<false, 1><<<pg, 256, 0, stream>>>(hidden, Wk, bk, t_k, k_up, kpad);
  lora_proj_kernel<false, 2><<<pg, 256, 0, stream>>>(hidden, Wv, bv, t_v, v_up, vpad);

  // 4. flash attention
  attn_kernel<<<dim3(SEQ/128, BATCH*HEADS), 256, 0, stream>>>(qpad, kpad, vpad, attnbuf);

  // 5. LoRA down for output projection (f16 source)
  lora_down_kernel<true><<<M_TOTAL/64, 256, 0, stream>>>(attnbuf, o_down, t_o);

  // 6. output projection (+ fused LoRA), dense f32 result
  lora_proj_kernel<true, 0><<<pg, 256, 0, stream>>>(attnbuf, Wo, bo, t_o, o_up, out);
}
